// OptimizedHyperMoENeck_6064493822107
// MI455X (gfx1250) — matrix operand order repaired and checked by reading, not compiled, not run on hardware
//
#include <hip/hip_runtime.h>
#include <hip/hip_bf16.h>
#include <math.h>

typedef __attribute__((ext_vector_type(16))) _Float16 v16h;
typedef __attribute__((ext_vector_type(8)))  _Float16 v8h;
typedef __attribute__((ext_vector_type(8)))  float    v8f;

#define BB   8
#define CC   256
#define HH   64
#define WWD  64
#define HWP  (HH*WWD)          // 4096
#define NTOT (BB*CC*HWP)       // 8388608
#define NEXP 4
#define NDOM 4
#define CR   16

// ---------------------------------------------------------------------------
// Weight pre-pack: OIHW f32 -> per-lane WMMA-A-fragment f16 streams.
// apack layout: [conv(16)][t(9)][kc(8)][co_tile(16)][lane(32)][i(16)] f16
//   lane l holds M = co_tile*16 + (l%16); K-base = kc*32 + 16*(l/16);
//   element i -> K = base+i   (contiguous, matches B fragment)
// ---------------------------------------------------------------------------
__global__ __launch_bounds__(256)
void pack_weights(const float* __restrict__ dt_w1, const float* __restrict__ dt_w2,
                  const float* __restrict__ c1_w,  const float* __restrict__ c2_w,
                  _Float16* __restrict__ apack)
{
    const size_t per_conv = 9ull * 8 * 16 * 32 * 16;   // 589824
    size_t gid = (size_t)blockIdx.x * 256 + threadIdx.x;
    if (gid >= 16 * per_conv) return;
    int conv = (int)(gid / per_conv);
    size_t rem = gid % per_conv;
    int i    = (int)(rem & 15);  rem >>= 4;
    int lane = (int)(rem & 31);  rem >>= 5;
    int cot  = (int)(rem & 15);  rem >>= 4;
    int kc   = (int)(rem & 7);   rem >>= 3;
    int t    = (int)rem;                         // 0..8 (kh*3+kw)
    int base = 16 * (lane >> 4);
    int kk   = kc * 32 + base + i;   // ci
    int co   = cot * 16 + (lane & 15);
    int e = conv >> 2, which = conv & 3;
    const float* w = (which == 0) ? dt_w1 : (which == 1) ? dt_w2
                   : (which == 2) ? c1_w  : c2_w;     // all (E,C,C,3,3)
    float v = w[(((size_t)e * CC + co) * CC + kk) * 9 + t];
    apack[gid] = (_Float16)v;
}

__global__ __launch_bounds__(256)
void f32_to_f16(const float* __restrict__ in, _Float16* __restrict__ out, int n)
{
    int i = blockIdx.x * 256 + threadIdx.x;
    if (i < n) out[i] = (_Float16)in[i];
}

// ---------------------------------------------------------------------------
// conv3x3 (pad=1, 256->256) as implicit GEMM via v_wmma_f32_16x16x32_f16.
// grid = (512, 2)  block = 256 (8 wave32)
//   blockIdx.x : n-tile = (b, h0) -> one 64-wide image row
//   blockIdx.y : M half; wave w owns co_tile = y*8+w (16 output channels)
//
// K pipeline (double-buffered):
//   per kc (K chunk of 32 input channels): stage rows h0-1,h0,h0+1 (3 tiles,
//   [n][k] transposed, 12 KB) into LDS buffer p=kc&1; ONE barrier; then issue
//   the global loads for kc+1 into VGPRs (latency hidden under compute) and
//   run 36 WMMAs (9 taps x 4 N-subtiles) out of buffer p.  Buffer reuse is
//   fenced by the barrier of the intervening iteration (waits DScnt for all
//   waves).
//   Border rows: the LOAD ADDRESS is clamped to row h0 (block-uniform offset
//   select, no per-store branches -> all 24 ds_store_b16 unconditional); the
//   garbage tile is never read because the compute stage skips the dh tap
//   group (block-uniform), which implements the zero padding.
//   dw shifts are applied at the fragment read (k-contiguous ->
//   ds_load_b128 pairs); only literal edge lanes need a zero fragment.
// act: 0 = none, 1 = gelu(tanh), 2 = relu.  Fuses BN scale/bias (+conv bias).
// ---------------------------------------------------------------------------
__global__ __launch_bounds__(256)
void conv3x3_wmma(const _Float16* __restrict__ inf16,
                  const _Float16* __restrict__ apack,   // this conv's pack
                  const float* __restrict__ bscale, const float* __restrict__ bbias,
                  const float* __restrict__ cbias,      // may be nullptr
                  int act,
                  float* __restrict__ outf32,           // may be nullptr
                  _Float16* __restrict__ outf16)        // may be nullptr
{
    __shared__ __align__(32) _Float16 ldsB[2 * 3 * 64 * 32];   // 24 KB, [p][dh][n][k]

    const int tid  = threadIdx.x;
    const int wave = tid >> 5;
    const int lane = tid & 31;
    const int ntile = blockIdx.x;          // 0..511
    const int b  = ntile >> 6;
    const int h0 = ntile & 63;
    const int co_tile = blockIdx.y * 8 + wave;

    v8f acc[4];
#pragma unroll
    for (int s = 0; s < 4; ++s)
#pragma unroll
        for (int r = 0; r < 8; ++r) acc[s][r] = 0.0f;

    // staging map: one 16B global load per thread (8 f16 along w, fixed ci)
    const int sk  = tid >> 3;              // staged k   (0..31)
    const int sn8 = (tid & 7) * 8;         // staged n base (0,8,...,56)
    // WMMA fragment map
    const int khalf = (lane >> 4) * 16;    // B: K half per lane
    const int ncol  = lane & 15;           // B/C: N column

    const bool rk0 = (h0 > 0);             // row h0-1 valid  (block-uniform)
    const bool rk2 = (h0 < HH - 1);        // row h0+1 valid  (block-uniform)
    const int  off0 = rk0 ? -WWD : 0;      // clamped neighbor-row offsets
    const int  off2 = rk2 ?  WWD : 0;      // (invalid -> reload row h0; unused)

    v16h vzero;
#pragma unroll
    for (int i = 0; i < 16; ++i) vzero[i] = (_Float16)0.0f;

    // per-thread global base for staging (row h0, channel sk of current chunk)
    const _Float16* gbase0 = inf16 + ((size_t)(b * CC + sk) * HH + h0) * WWD + sn8;

    // ---- prologue: fetch rows for kc = 0 (clamped, unconditional) ----
    v8h reg0 = *(const v8h*)(gbase0 + off0);
    v8h reg1 = *(const v8h*)gbase0;
    v8h reg2 = *(const v8h*)(gbase0 + off2);

    for (int kc = 0; kc < 8; ++kc) {
        _Float16* buf = ldsB + (size_t)(kc & 1) * (3 * 64 * 32);
        // ---- store staged rows (transpose scatter into [n][k]), no branches ----
#pragma unroll
        for (int j = 0; j < 8; ++j) {
            const int a = (sn8 + j) * 32 + sk;
            buf[0 * 2048 + a] = reg0[j];
            buf[1 * 2048 + a] = reg1[j];
            buf[2 * 2048 + a] = reg2[j];
        }
        __syncthreads();                       // one barrier per K chunk

        // ---- issue next chunk's global loads (hide under WMMAs below) ----
        if (kc < 7) {
            const _Float16* gb = gbase0 + (size_t)(kc + 1) * 32 * HWP;
            reg0 = *(const v8h*)(gb + off0);
            reg1 = *(const v8h*)gb;
            reg2 = *(const v8h*)(gb + off2);
        }

        // ---- 9 taps x 4 N-subtiles = 36 WMMAs from buffer p ----
#pragma unroll
        for (int dhi = 0; dhi < 3; ++dhi) {
            if (dhi == 0 && !rk0) continue;    // block-uniform skip (zero pad)
            if (dhi == 2 && !rk2) continue;
            const _Float16* tile = buf + dhi * 2048;
#pragma unroll
            for (int dwi = 0; dwi < 3; ++dwi) {
                const int t = dhi * 3 + dwi;
                const _Float16* ap = apack +
                    ((((size_t)t * 8 + kc) * 16 + co_tile) * 32 + lane) * 16;
                v16h afrag = *(const v16h*)ap;
                __builtin_prefetch(ap + 8 * 16 * 32 * 16, 0, 1);  // next tap

#pragma unroll
                for (int s = 0; s < 4; ++s) {
                    const int ncl = s * 16 + ncol + dwi - 1;      // shifted col
                    v16h bfrag = (ncl >= 0 && ncl < 64)
                        ? *(const v16h*)&tile[ncl * 32 + khalf]
                        : vzero;
                    acc[s] = __builtin_amdgcn_wmma_f32_16x16x32_f16(
                        false, afrag, false, bfrag, (short)0, acc[s], false, false);
                }
            }
        }
    }

    // ---- epilogue: BN(+bias), activation, store f32/f16 ----
    const int mrow_off = (lane >> 4) * 8;
#pragma unroll
    for (int s = 0; s < 4; ++s) {
#pragma unroll
        for (int r = 0; r < 8; ++r) {
            const int co = co_tile * 16 + mrow_off + r;
            const int n  = s * 16 + ncol;
            float v  = acc[s][r];
            float sc = bscale[co];
            float bi = bbias[co];
            if (cbias) bi += sc * cbias[co];
            v = v * sc + bi;
            if (act == 1) {                          // gelu (tanh approx, JAX default)
                float x3 = v * v * v;
                v = 0.5f * v * (1.0f + tanhf(0.7978845608028654f * (v + 0.044715f * x3)));
            } else if (act == 2) {
                v = fmaxf(v, 0.0f);
            }
            const size_t idx = ((size_t)(b * CC + co) * HH + h0) * WWD + n;
            if (outf32) outf32[idx] = v;
            if (outf16) outf16[idx] = (_Float16)v;
        }
    }
}

// ---------------------------------------------------------------------------
// per-(b,c) spatial mean & max.  grid = B*C, block = 256
// ---------------------------------------------------------------------------
__global__ __launch_bounds__(256)
void stats_mean_max(const float* __restrict__ in, float* __restrict__ mean,
                    float* __restrict__ mx)
{
    __shared__ float ssum[256], smax[256];
    const int bc = blockIdx.x;
    const float* p = in + (size_t)bc * HWP;
    float s = 0.0f, m = -INFINITY;
    for (int i = threadIdx.x; i < HWP; i += 256) {
        float v = p[i]; s += v; m = fmaxf(m, v);
    }
    ssum[threadIdx.x] = s; smax[threadIdx.x] = m;
    __syncthreads();
    for (int st = 128; st > 0; st >>= 1) {
        if (threadIdx.x < st) {
            ssum[threadIdx.x] += ssum[threadIdx.x + st];
            smax[threadIdx.x] = fmaxf(smax[threadIdx.x], smax[threadIdx.x + st]);
        }
        __syncthreads();
    }
    if (threadIdx.x == 0) { mean[bc] = ssum[0] * (1.0f / HWP); mx[bc] = smax[0]; }
}

// gate FC1: gi=[gap|gmp|dom_emb] @ g_w1 + g_b1.  grid=B, block=256
__global__ __launch_bounds__(256)
void gate_fc1(const float* __restrict__ gap, const float* __restrict__ gmp,
              const float* __restrict__ dom_emb, const int* __restrict__ dom_ids,
              const float* __restrict__ w1, const float* __restrict__ b1,
              float* __restrict__ hg)
{
    const int b = blockIdx.x, co = threadIdx.x;
    float s = b1[co];
    for (int i = 0; i < CC; ++i) s += gap[b * CC + i] * w1[i * CC + co];
    for (int i = 0; i < CC; ++i) s += gmp[b * CC + i] * w1[(CC + i) * CC + co];
    const float* de = dom_emb + dom_ids[b] * 16;
    for (int i = 0; i < 16; ++i)  s += de[i] * w1[(2 * CC + i) * CC + co];
    hg[b * CC + co] = s;
}

// LayerNorm -> ReLU -> FC2 -> softmax -> top-2 renorm.  grid=B, block=256
__global__ __launch_bounds__(256)
void gate_ln_topk(const float* __restrict__ hg, const float* __restrict__ lns,
                  const float* __restrict__ lnb, const float* __restrict__ w2,
                  const float* __restrict__ b2, float* __restrict__ wfull)
{
    __shared__ float red[256];
    __shared__ float a[256];
    __shared__ float logits[NEXP];
    const int b = blockIdx.x, t = threadIdx.x;
    float v = hg[b * CC + t];
    red[t] = v; __syncthreads();
    for (int st = 128; st > 0; st >>= 1) { if (t < st) red[t] += red[t + st]; __syncthreads(); }
    float mu = red[0] * (1.0f / CC); __syncthreads();
    float dv = v - mu; red[t] = dv * dv; __syncthreads();
    for (int st = 128; st > 0; st >>= 1) { if (t < st) red[t] += red[t + st]; __syncthreads(); }
    float var = red[0] * (1.0f / CC);
    a[t] = fmaxf((v - mu) * rsqrtf(var + 1e-5f) * lns[t] + lnb[t], 0.0f);
    __syncthreads();
    if (t < NEXP) {
        float s = b2[t];
        for (int c = 0; c < CC; ++c) s += a[c] * w2[c * NEXP + t];
        logits[t] = s;
    }
    __syncthreads();
    if (t == 0) {
        float p[NEXP];
        float mxl = logits[0];
        for (int e = 1; e < NEXP; ++e) mxl = fmaxf(mxl, logits[e]);
        float sum = 0.0f;
        for (int e = 0; e < NEXP; ++e) { p[e] = __expf(logits[e] - mxl); sum += p[e]; }
        for (int e = 0; e < NEXP; ++e) p[e] /= sum;
        int i0 = 0;
        for (int e = 1; e < NEXP; ++e) if (p[e] > p[i0]) i0 = e;
        int i1 = -1;
        for (int e = 0; e < NEXP; ++e) if (e != i0 && (i1 < 0 || p[e] > p[i1])) i1 = e;
        float wk = p[i0] + p[i1];
        float o[NEXP] = {0, 0, 0, 0};
        o[i0] = p[i0] / wk; o[i1] = p[i1] / wk;
        for (int e = 0; e < NEXP; ++e) wfull[b * NEXP + e] = o[e];
    }
}

// domain gate: g1[b] = sigmoid(dmean[b,:] . gw + gb[e]).  grid=B, block=256
__global__ __launch_bounds__(256)
void domain_gate(const float* __restrict__ dmean, const float* __restrict__ gw,
                 const float* __restrict__ gb, int e, float* __restrict__ g1)
{
    __shared__ float red[256];
    const int b = blockIdx.x, t = threadIdx.x;
    red[t] = dmean[b * CC + t] * gw[t];
    __syncthreads();
    for (int st = 128; st > 0; st >>= 1) { if (t < st) red[t] += red[t + st]; __syncthreads(); }
    if (t == 0) g1[b] = 1.0f / (1.0f + __expf(-(red[0] + gb[e])));
}

// out = x + g1[b]*d ; emit f32 and f16 copies
__global__ __launch_bounds__(256)
void residual_out(const float* __restrict__ x, const float* __restrict__ d,
                  const float* __restrict__ g1, float* __restrict__ of32,
                  _Float16* __restrict__ of16, int n)
{
    int i = blockIdx.x * 256 + threadIdx.x;
    if (i >= n) return;
    int b = i >> 20;                        // CC*HWP = 1<<20 per batch
    float v = x[i] + g1[b] * d[i];
    of32[i] = v; of16[i] = (_Float16)v;
}

// channel attention (shared MLP on avg & max).  grid=B, block=256
__global__ __launch_bounds__(256)
void channel_att(const float* __restrict__ havg, const float* __restrict__ hmax,
                 const float* __restrict__ w1, const float* __restrict__ w2,
                 float* __restrict__ ca)
{
    __shared__ float hid[CR];
    const int b = blockIdx.x, t = threadIdx.x;
    if (t < CR) {
        float sa = 0.0f, sm = 0.0f;
        for (int c = 0; c < CC; ++c) {
            float w = w1[t * CC + c];
            sa += w * havg[b * CC + c];
            sm += w * hmax[b * CC + c];
        }
        hid[t] = fmaxf(sa, 0.0f) + fmaxf(sm, 0.0f);
    }
    __syncthreads();
    float s = 0.0f;
    for (int r = 0; r < CR; ++r) s += w2[t * CR + r] * hid[r];
    ca[b * CC + t] = 1.0f / (1.0f + __expf(-s));
}

// per-pixel channel mean/max of h3*ca -> planes[B][2][HW]
__global__ __launch_bounds__(256)
void spatial_stats(const float* __restrict__ h3, const float* __restrict__ ca,
                   float* __restrict__ planes)
{
    int i = blockIdx.x * 256 + threadIdx.x;
    if (i >= BB * HWP) return;
    int b = i >> 12, p = i & (HWP - 1);
    const float* hp = h3 + (size_t)b * CC * HWP + p;
    const float* cp = ca + b * CC;
    float s = 0.0f, m = -INFINITY;
    for (int c = 0; c < CC; ++c) {
        float v = hp[(size_t)c * HWP] * cp[c];
        s += v; m = fmaxf(m, v);
    }
    planes[(b * 2 + 0) * HWP + p] = s * (1.0f / CC);
    planes[(b * 2 + 1) * HWP + p] = m;
}

// 7x7 conv (2->1, pad 3) + sigmoid
__global__ __launch_bounds__(256)
void spatial_conv(const float* __restrict__ planes, const float* __restrict__ sw,
                  float* __restrict__ sa)
{
    int i = blockIdx.x * 256 + threadIdx.x;
    if (i >= BB * HWP) return;
    int b = i >> 12, p = i & (HWP - 1), h = p >> 6, w = p & 63;
    float s = 0.0f;
    for (int ch = 0; ch < 2; ++ch)
        for (int kh = 0; kh < 7; ++kh) {
            int hy = h + kh - 3; if (hy < 0 || hy >= HH) continue;
            for (int kw = 0; kw < 7; ++kw) {
                int wx = w + kw - 3; if (wx < 0 || wx >= WWD) continue;
                s += sw[(ch * 7 + kh) * 7 + kw] * planes[(b * 2 + ch) * HWP + hy * 64 + wx];
            }
        }
    sa[i] = 1.0f / (1.0f + __expf(-s));
}

// he = relu(h3*ca*sa + resid) + adapter[dom[b]]; out (+)= wfull[b,e]*he
__global__ __launch_bounds__(256)
void final_accum(const float* __restrict__ h3, const float* __restrict__ ca,
                 const float* __restrict__ sa, const float* __restrict__ resid,
                 const float* __restrict__ adapter, const int* __restrict__ dom_ids,
                 const float* __restrict__ wfull, int e, int first,
                 float* __restrict__ out, int n)
{
    int i = blockIdx.x * 256 + threadIdx.x;
    if (i >= n) return;
    int b = i >> 20;
    int c = (i >> 12) & (CC - 1);
    int p = i & (HWP - 1);
    float he = fmaxf(h3[i] * ca[b * CC + c] * sa[b * HWP + p] + resid[i], 0.0f)
             + adapter[(size_t)dom_ids[b] * CC + c];
    float wv = wfull[b * NEXP + e];
    if (first) out[i] = wv * he;
    else       out[i] += wv * he;
}

// ---------------------------------------------------------------------------
extern "C" void kernel_launch(void* const* d_in, const int* in_sizes, int n_in,
                              void* d_out, int out_size, void* d_ws, size_t ws_size,
                              hipStream_t stream)
{
    const float* x        = (const float*)d_in[0];
    const int*   dom_ids  = (const int*)  d_in[1];
    const float* g_w1     = (const float*)d_in[2];
    const float* g_b1     = (const float*)d_in[3];
    const float* ln_s     = (const float*)d_in[4];
    const float* ln_b     = (const float*)d_in[5];
    const float* g_w2     = (const float*)d_in[6];
    const float* g_b2     = (const float*)d_in[7];
    const float* dom_emb  = (const float*)d_in[8];
    const float* e_dt_w1  = (const float*)d_in[9];
    const float* e_dt_bn1s= (const float*)d_in[10];
    const float* e_dt_bn1b= (const float*)d_in[11];
    const float* e_dt_w2  = (const float*)d_in[12];
    const float* e_dt_bn2s= (const float*)d_in[13];
    const float* e_dt_bn2b= (const float*)d_in[14];
    const float* e_gate_w = (const float*)d_in[15];
    const float* e_gate_b = (const float*)d_in[16];
    const float* e_c1_w   = (const float*)d_in[17];
    const float* e_c1_b   = (const float*)d_in[18];
    const float* e_bn1s   = (const float*)d_in[19];
    const float* e_bn1b   = (const float*)d_in[20];
    const float* e_c2_w   = (const float*)d_in[21];
    const float* e_c2_b   = (const float*)d_in[22];
    const float* e_bn2s   = (const float*)d_in[23];
    const float* e_bn2b   = (const float*)d_in[24];
    const float* e_ca_w1  = (const float*)d_in[25];
    const float* e_ca_w2  = (const float*)d_in[26];
    const float* e_sa_w   = (const float*)d_in[27];
    const float* e_adapter= (const float*)d_in[28];
    float* out = (float*)d_out;

    // ---- workspace carve-out ----
    char* ws = (char*)d_ws;
    size_t off = 0;
    auto alloc = [&](size_t bytes) -> void* {
        void* p = ws + off;
        off += (bytes + 255) & ~(size_t)255;
        return p;
    };
    const size_t per_conv = 9ull * 8 * 16 * 32 * 16;          // 589824 f16
    _Float16* apack = (_Float16*)alloc(16 * per_conv * 2);    // 18.0 MB
    _Float16* xf16  = (_Float16*)alloc((size_t)NTOT * 2);     // 16 MB
    _Float16* tf16a = (_Float16*)alloc((size_t)NTOT * 2);     // h / h2
    _Float16* tf16b = (_Float16*)alloc((size_t)NTOT * 2);     // out (f16)
    float* d32   = (float*)alloc((size_t)NTOT * 4);           // d, then h3
    float* out32 = (float*)alloc((size_t)NTOT * 4);           // residual "out"
    float* gap   = (float*)alloc(BB * CC * 4);
    float* gmp   = (float*)alloc(BB * CC * 4);
    float* hg    = (float*)alloc(BB * CC * 4);
    float* wfull = (float*)alloc(BB * NEXP * 4);
    float* dmean = (float*)alloc(BB * CC * 4);
    float* dmax  = (float*)alloc(BB * CC * 4);
    float* havg  = (float*)alloc(BB * CC * 4);
    float* hmax  = (float*)alloc(BB * CC * 4);
    float* g1    = (float*)alloc(BB * 4);
    float* ca    = (float*)alloc(BB * CC * 4);
    float* planes= (float*)alloc((size_t)BB * 2 * HWP * 4);
    float* sa    = (float*)alloc((size_t)BB * HWP * 4);
    (void)ws_size; (void)n_in; (void)in_sizes; (void)out_size;

    const int eb = (NTOT + 255) / 256;          // element-wise grid
    const dim3 cgrid(512, 2);                   // conv grid: 512 n-tiles x 2 M halves

    // ---- one-time prep ----
    pack_weights<<<(int)((16 * per_conv + 255) / 256), 256, 0, stream>>>(
        e_dt_w1, e_dt_w2, e_c1_w, e_c2_w, apack);
    f32_to_f16<<<eb, 256, 0, stream>>>(x, xf16, NTOT);

    // ---- gating ----
    stats_mean_max<<<BB * CC, 256, 0, stream>>>(x, gap, gmp);
    gate_fc1<<<BB, 256, 0, stream>>>(gap, gmp, dom_emb, dom_ids, g_w1, g_b1, hg);
    gate_ln_topk<<<BB, 256, 0, stream>>>(hg, ln_s, ln_b, g_w2, g_b2, wfull);

    // ---- experts ----
    for (int e = 0; e < NEXP; ++e) {
        const _Float16* ap = apack + (size_t)e * 4 * per_conv;

        // h = gelu(bn1(conv(x, dt_w1)))                        -> f16
        conv3x3_wmma<<<cgrid, 256, 0, stream>>>(
            xf16, ap + 0 * per_conv, e_dt_bn1s + e * CC, e_dt_bn1b + e * CC,
            nullptr, 1, nullptr, tf16a);
        // d = bn2(conv(h, dt_w2))                              -> f32
        conv3x3_wmma<<<cgrid, 256, 0, stream>>>(
            tf16a, ap + 1 * per_conv, e_dt_bn2s + e * CC, e_dt_bn2b + e * CC,
            nullptr, 0, d32, nullptr);

        // domain gate + residual: out = x + g1*d
        stats_mean_max<<<BB * CC, 256, 0, stream>>>(d32, dmean, dmax);
        domain_gate<<<BB, 256, 0, stream>>>(dmean, e_gate_w + e * CC, e_gate_b, e, g1);
        residual_out<<<eb, 256, 0, stream>>>(x, d32, g1, out32, tf16b, NTOT);

        // h2 = relu(bn1(conv(out, c1_w) + c1_b))               -> f16 (reuse tf16a)
        conv3x3_wmma<<<cgrid, 256, 0, stream>>>(
            tf16b, ap + 2 * per_conv, e_bn1s + e * CC, e_bn1b + e * CC,
            e_c1_b + e * CC, 2, nullptr, tf16a);
        // h3 = bn2(conv(h2, c2_w) + c2_b)                      -> f32 (reuse d32)
        conv3x3_wmma<<<cgrid, 256, 0, stream>>>(
            tf16a, ap + 3 * per_conv, e_bn2s + e * CC, e_bn2b + e * CC,
            e_c2_b + e * CC, 0, d32, nullptr);

        // CBAM channel + spatial attention, adapter, weighted accumulate
        stats_mean_max<<<BB * CC, 256, 0, stream>>>(d32, havg, hmax);
        channel_att<<<BB, 256, 0, stream>>>(havg, hmax,
            e_ca_w1 + (size_t)e * CR * CC, e_ca_w2 + (size_t)e * CC * CR, ca);
        spatial_stats<<<(BB * HWP + 255) / 256, 256, 0, stream>>>(d32, ca, planes);
        spatial_conv<<<(BB * HWP + 255) / 256, 256, 0, stream>>>(
            planes, e_sa_w + (size_t)e * 2 * 49, sa);
        final_accum<<<eb, 256, 0, stream>>>(d32, ca, sa, out32,
            e_adapter + (size_t)e * NDOM * CC, dom_ids, wfull, e, (e == 0) ? 1 : 0,
            out, NTOT);
    }
}
